// DynamicLinkPredictor_78357383348231
// MI455X (gfx1250) — matrix operand order repaired and checked by reading, not compiled, not run on hardware
//
#include <hip/hip_runtime.h>
#include <hip/hip_bf16.h>

#define NN  2048
#define FIN 64
#define HH  32

typedef __attribute__((ext_vector_type(16))) _Float16 v16h;
typedef __attribute__((ext_vector_type(8)))  float    v8f;

// ---------------------------------------------------------------------------
// WMMA fragment builders (wave32; A and B both use contiguous k = 16*(lane/16) + e)
// ---------------------------------------------------------------------------
__device__ __forceinline__ v16h frag_A_f32(const float* __restrict__ src, int ld,
                                           int row0, int k0, int lane) {
  const int r  = row0 + (lane & 15);
  const int hi = (lane >> 4) & 1;
  const float* p = src + r * ld + k0;
  v16h a;
#pragma unroll
  for (int e = 0; e < 16; ++e) {
    const int k = e + 16 * hi;
    a[e] = (_Float16)p[k];
  }
  return a;
}

__device__ __forceinline__ v16h frag_B_f32(const float* __restrict__ w, int ldn,
                                           int k0, int n0, int lane) {
  const int col = n0 + (lane & 15);
  const int hi  = (lane >> 4) & 1;
  v16h b;
#pragma unroll
  for (int e = 0; e < 16; ++e) {
    const int k = k0 + e + 16 * hi;
    b[e] = (_Float16)w[k * ldn + col];
  }
  return b;
}

__device__ __forceinline__ v8f wmma16(v16h a, v16h b, v8f c) {
  return __builtin_amdgcn_wmma_f32_16x16x32_f16(false, a, false, b,
                                                (short)0, c, false, false);
}

__device__ __forceinline__ float sigm(float x) { return 1.0f / (1.0f + __expf(-x)); }

// ---------------------------------------------------------------------------
// CDNA5 async global->LDS copies (GLOBAL_LOAD_ASYNC_TO_LDS_*, ASYNCcnt).
// Low 32 bits of a flat LDS pointer are the LDS byte address (ISA 10.2).
// ---------------------------------------------------------------------------
__device__ __forceinline__ void async_ld_b128(void* lds, const void* gptr) {
  unsigned lo = (unsigned)(unsigned long long)(uintptr_t)lds;
  asm volatile("global_load_async_to_lds_b128 %0, %1, off"
               :: "v"(lo), "v"(gptr) : "memory");
}
__device__ __forceinline__ void async_ld_b32(void* lds, const void* gptr) {
  unsigned lo = (unsigned)(unsigned long long)(uintptr_t)lds;
  asm volatile("global_load_async_to_lds_b32 %0, %1, off"
               :: "v"(lo), "v"(gptr) : "memory");
}
__device__ __forceinline__ void wait_async0() {
  asm volatile("s_wait_asynccnt 0" ::: "memory");
}

// ---------------------------------------------------------------------------
// GCLSTM layer with zero incoming state: Cheb term degenerates to bias bc_g.
// 3 waves = gates i/c/o; two 16x16 WMMA tiles each over K; combine via LDS.
// ---------------------------------------------------------------------------
template <int CIN>
__global__ __launch_bounds__(96)
void gclstm_layer_kernel(const float* __restrict__ hin,
                         const float* __restrict__ Wi, const float* __restrict__ Wc,
                         const float* __restrict__ Wo,
                         const float* __restrict__ bi, const float* __restrict__ bc,
                         const float* __restrict__ bo,
                         const float* __restrict__ bci, const float* __restrict__ bcc,
                         const float* __restrict__ bco,
                         float* __restrict__ hout) {
  __shared__ float g[3][16][HH];
  const int lane = threadIdx.x & 31;
  const int w    = threadIdx.x >> 5;          // 0:i  1:c  2:o
  const int row0 = blockIdx.x * 16;

  const float* W   = (w == 0) ? Wi  : (w == 1) ? Wc  : Wo;
  const float* bL  = (w == 0) ? bi  : (w == 1) ? bc  : bo;
  const float* bCh = (w == 0) ? bci : (w == 1) ? bcc : bco;

  v8f acc0 = {};
  v8f acc1 = {};
#pragma unroll
  for (int kb = 0; kb < CIN; kb += 32) {
    v16h af = frag_A_f32(hin, CIN, row0, kb, lane);
    v16h b0 = frag_B_f32(W, HH, kb, 0,  lane);
    v16h b1 = frag_B_f32(W, HH, kb, 16, lane);
    acc0 = wmma16(af, b0, acc0);
    acc1 = wmma16(af, b1, acc1);
  }

  const int col = lane & 15;
  const int hi  = (lane >> 4) & 1;
  const float bs0 = bL[col]      + bCh[col];
  const float bs1 = bL[col + 16] + bCh[col + 16];
#pragma unroll
  for (int v = 0; v < 8; ++v) {
    float x0 = acc0[v] + bs0;
    float x1 = acc1[v] + bs1;
    if (w == 1) { x0 = tanhf(x0); x1 = tanhf(x1); }
    else        { x0 = sigm(x0);  x1 = sigm(x1);  }
    g[w][v + 8 * hi][col]      = x0;
    g[w][v + 8 * hi][col + 16] = x1;
  }
  __syncthreads();

  // h = O * tanh(I * T)    (forget gate * C_prev == 0, dropped)
  for (int idx = threadIdx.x; idx < 16 * HH; idx += 96) {
    const int r = idx >> 5, cc = idx & (HH - 1);
    const float I = g[0][r][cc], T = g[1][r][cc], O = g[2][r][cc];
    hout[(row0 + r) * HH + cc] = O * tanhf(I * T);
  }
}

// ---------------------------------------------------------------------------
// Head projections: a = h@W1[:32] + b1 (wave0), c = h@W1[32:] (wave1).
// ---------------------------------------------------------------------------
__global__ __launch_bounds__(64)
void head_proj_kernel(const float* __restrict__ h, const float* __restrict__ W1,
                      const float* __restrict__ b1,
                      float* __restrict__ aout, float* __restrict__ cout) {
  const int lane = threadIdx.x & 31;
  const int w    = threadIdx.x >> 5;
  const int row0 = blockIdx.x * 16;
  const float* W = W1 + w * HH * HH;
  float* out = (w == 0) ? aout : cout;

  v16h af  = frag_A_f32(h, HH, row0, 0, lane);
  v16h b0  = frag_B_f32(W, HH, 0, 0,  lane);
  v16h b1f = frag_B_f32(W, HH, 0, 16, lane);
  v8f acc0 = {}, acc1 = {};
  acc0 = wmma16(af, b0, acc0);
  acc1 = wmma16(af, b1f, acc1);

  const int col = lane & 15;
  const int hi  = (lane >> 4) & 1;
  const float add0 = (w == 0) ? b1[col]      : 0.0f;
  const float add1 = (w == 0) ? b1[col + 16] : 0.0f;
#pragma unroll
  for (int v = 0; v < 8; ++v) {
    const int r = row0 + v + 8 * hi;
    out[r * HH + col]      = acc0[v] + add0;
    out[r * HH + col + 16] = acc1[v] + add1;
  }
}

// ---------------------------------------------------------------------------
// Pairwise head: adj[i][j] = sigmoid(sum_h relu(a[i][h]+c[j][h]) * w2[h] + b2).
// Tiles staged via CDNA5 async global->LDS DMA (ASYNCcnt), then VALU dot.
// ---------------------------------------------------------------------------
__global__ __launch_bounds__(256)
void pairwise_head_kernel(const float* __restrict__ a, const float* __restrict__ c,
                          const float* __restrict__ w2, const float* __restrict__ b2,
                          float* __restrict__ adj) {
  __shared__ float sa[16][HH + 1];
  __shared__ float sc[16][HH + 1];
  __shared__ float sw[HH];
  const int t  = threadIdx.x;
  const int bi = blockIdx.y, bj = blockIdx.x;
  const int wv = t >> 5;
  const int l  = t & 31;

  if (wv == 0) {                 // a-tile: 4 issues x 32 lanes x 16B = 2 KB
    const int rbase = l >> 3, seg = l & 7;
#pragma unroll
    for (int rr = 0; rr < 4; ++rr) {
      const int row = rr * 4 + rbase;
      async_ld_b128(&sa[row][seg * 4],
                    &a[(size_t)(bi * 16 + row) * HH + seg * 4]);
    }
    wait_async0();
  } else if (wv == 1) {          // c-tile
    const int rbase = l >> 3, seg = l & 7;
#pragma unroll
    for (int rr = 0; rr < 4; ++rr) {
      const int row = rr * 4 + rbase;
      async_ld_b128(&sc[row][seg * 4],
                    &c[(size_t)(bj * 16 + row) * HH + seg * 4]);
    }
    wait_async0();
  } else if (wv == 2) {          // w2 vector (128 B)
    async_ld_b32(&sw[l], &w2[l]);
    wait_async0();
  }
  __syncthreads();

  const int ti = t >> 4, tj = t & 15;
  float s = b2[0];
#pragma unroll
  for (int hch = 0; hch < HH; ++hch)
    s = fmaf(fmaxf(sa[ti][hch] + sc[tj][hch], 0.0f), sw[hch], s);

  adj[(size_t)(bi * 16 + ti) * NN + (bj * 16 + tj)] = 1.0f / (1.0f + __expf(-s));
}

// ---------------------------------------------------------------------------
// Launch. Param flattening assumes sorted-key dict order:
//  0:x  1:edge_index  2:edge_weight  (edges/Wc_* dead: state==0)
//  per layer (base 3 / 19): W_c,W_f,W_i,W_o, Wc_c..Wc_o, b_c,b_f,b_i,b_o,
//                           bc_c,bc_f,bc_i,bc_o
//  35:W1  36:W2  37:b1  38:b2
// ---------------------------------------------------------------------------
extern "C" void kernel_launch(void* const* d_in, const int* in_sizes, int n_in,
                              void* d_out, int out_size, void* d_ws, size_t ws_size,
                              hipStream_t stream) {
  if (n_in < 39) return;
  const float* x = (const float*)d_in[0];

  const int L0 = 3, L1 = 19;
  const float* Wi0  = (const float*)d_in[L0 + 2];
  const float* Wc0  = (const float*)d_in[L0 + 0];
  const float* Wo0  = (const float*)d_in[L0 + 3];
  const float* bi0  = (const float*)d_in[L0 + 10];
  const float* bc0  = (const float*)d_in[L0 + 8];
  const float* bo0  = (const float*)d_in[L0 + 11];
  const float* bci0 = (const float*)d_in[L0 + 14];
  const float* bcc0 = (const float*)d_in[L0 + 12];
  const float* bco0 = (const float*)d_in[L0 + 15];

  const float* Wi1  = (const float*)d_in[L1 + 2];
  const float* Wc1  = (const float*)d_in[L1 + 0];
  const float* Wo1  = (const float*)d_in[L1 + 3];
  const float* bi1  = (const float*)d_in[L1 + 10];
  const float* bc1  = (const float*)d_in[L1 + 8];
  const float* bo1  = (const float*)d_in[L1 + 11];
  const float* bci1 = (const float*)d_in[L1 + 14];
  const float* bcc1 = (const float*)d_in[L1 + 12];
  const float* bco1 = (const float*)d_in[L1 + 15];

  const float* W1 = (const float*)d_in[35];
  const float* W2 = (const float*)d_in[36];
  const float* b1 = (const float*)d_in[37];
  const float* b2 = (const float*)d_in[38];

  float* ws = (float*)d_ws;
  float* h1 = ws;
  float* h2 = ws + (size_t)NN * HH;
  float* av = ws + (size_t)2 * NN * HH;
  float* cv = ws + (size_t)3 * NN * HH;

  gclstm_layer_kernel<FIN><<<NN / 16, 96, 0, stream>>>(
      x, Wi0, Wc0, Wo0, bi0, bc0, bo0, bci0, bcc0, bco0, h1);
  gclstm_layer_kernel<HH><<<NN / 16, 96, 0, stream>>>(
      h1, Wi1, Wc1, Wo1, bi1, bc1, bo1, bci1, bcc1, bco1, h2);
  head_proj_kernel<<<NN / 16, 64, 0, stream>>>(h2, W1, b1, av, cv);
  pairwise_head_kernel<<<dim3(NN / 16, NN / 16), 256, 0, stream>>>(
      av, cv, W2, b2, (float*)d_out);
}
